// OTAD_NN_19464791786025
// MI455X (gfx1250) — matrix operand order repaired and checked by reading, not compiled, not run on hardware
//
#include <hip/hip_runtime.h>
#include <hip/hip_bf16.h>
#include <float.h>

typedef _Float16 v16h __attribute__((ext_vector_type(16)));
typedef _Float16 v8h  __attribute__((ext_vector_type(8)));
typedef float    v8f  __attribute__((ext_vector_type(8)));

#define B_       64
#define N_DML    6000
#define DF_      512
#define P_       24960      // GEMM K (24960 = 390 * 64)
#define DIM_     2048       // GEMM N
#define KNN_     10
#define HID_     384
#define TOK_     65

// ---------------------------------------------------------------------------
// Stage 1: cls = x @ W_dml   (f32 scalar; precision-sensitive retrieval path)
// ---------------------------------------------------------------------------
__global__ void cls_kernel(const float* __restrict__ x, const float* __restrict__ Wdml,
                           float* __restrict__ cls) {
    int idx = blockIdx.x * blockDim.x + threadIdx.x;   // 64*512
    if (idx >= B_ * DF_) return;
    int b = idx >> 9;
    int j = idx & 511;
    const float* xr = x + b * 3072;
    float acc = 0.f;
    for (int i = 0; i < 3072; ++i)
        acc += xr[i] * Wdml[i * DF_ + j];
    cls[idx] = acc;
}

__global__ void dml2_kernel(const float* __restrict__ dml, float* __restrict__ dml2) {
    int n = blockIdx.x * blockDim.x + threadIdx.x;
    if (n >= N_DML) return;
    const float* r = dml + (size_t)n * DF_;
    float acc = 0.f;
    for (int i = 0; i < DF_; ++i) acc += r[i] * r[i];
    dml2[n] = acc;
}

// score[b][n] = ||dml_n||^2 - 2 cls_b . dml_n   (monotone in true distance per row)
__global__ void score_kernel(const float* __restrict__ cls, const float* __restrict__ dml,
                             const float* __restrict__ dml2, float* __restrict__ score) {
    int n = blockIdx.x * blockDim.x + threadIdx.x;
    int b = blockIdx.y;
    if (n >= N_DML) return;
    const float* c = cls + b * DF_;
    const float* d = dml + (size_t)n * DF_;
    float acc = 0.f;
    for (int i = 0; i < DF_; ++i) acc += c[i] * d[i];
    score[(size_t)b * N_DML + n] = dml2[n] - 2.0f * acc;
}

// top-K smallest per row (iterative min + mask); score is ws, rewritten each call
__global__ void topk_kernel(float* __restrict__ score, int* __restrict__ idx_out) {
    int b = blockIdx.x;
    __shared__ float sval[256];
    __shared__ int   sidx[256];
    float* row = score + (size_t)b * N_DML;
    for (int k = 0; k < KNN_; ++k) {
        float best = FLT_MAX; int bi = 0x7FFFFFFF;
        for (int n = threadIdx.x; n < N_DML; n += 256) {
            float v = row[n];
            if (v < best || (v == best && n < bi)) { best = v; bi = n; }
        }
        sval[threadIdx.x] = best; sidx[threadIdx.x] = bi;
        __syncthreads();
        for (int s = 128; s > 0; s >>= 1) {
            if (threadIdx.x < s) {
                float ov = sval[threadIdx.x + s]; int oi = sidx[threadIdx.x + s];
                if (ov < sval[threadIdx.x] ||
                    (ov == sval[threadIdx.x] && oi < sidx[threadIdx.x])) {
                    sval[threadIdx.x] = ov; sidx[threadIdx.x] = oi;
                }
            }
            __syncthreads();
        }
        if (threadIdx.x == 0) { idx_out[b * KNN_ + k] = sidx[0]; row[sidx[0]] = FLT_MAX; }
        __syncthreads();
    }
}

// ---------------------------------------------------------------------------
// Patch embed: embed[b, t, h], t=0 -> cls token, t>0 -> 4x4 patch @ W_patch
// ---------------------------------------------------------------------------
__global__ void patch_embed_kernel(const float* __restrict__ x, const float* __restrict__ Wp,
                                   const float* __restrict__ cls_tok, float* __restrict__ embed) {
    int idx = blockIdx.x * blockDim.x + threadIdx.x;   // 64*65*384
    if (idx >= B_ * TOK_ * HID_) return;
    int b = idx / (TOK_ * HID_);
    int r = idx % (TOK_ * HID_);
    int t = r / HID_;
    int h = r % HID_;
    if (t == 0) { embed[idx] = cls_tok[h]; return; }
    const float mean[3]    = {0.4914f, 0.4822f, 0.4465f};
    const float inv_std[3] = {1.0f / 0.2470f, 1.0f / 0.2435f, 1.0f / 0.2616f};
    int p = t - 1;
    int py8 = p >> 3, px8 = p & 7;
    const float* xb = x + (size_t)b * 3072;
    float acc = 0.f;
    for (int c = 0; c < 3; ++c) {
        for (int py = 0; py < 4; ++py) {
            for (int px = 0; px < 4; ++px) {
                int f = c * 16 + py * 4 + px;
                float xv = xb[c * 1024 + (py8 * 4 + py) * 32 + (px8 * 4 + px)];
                acc += (xv - mean[c]) * inv_std[c] * Wp[f * HID_ + h];
            }
        }
    }
    embed[idx] = acc;
}

// ---------------------------------------------------------------------------
// Row-pointer tables: gather-by-indirection (avoids materializing 128MB)
// ---------------------------------------------------------------------------
__global__ void build_rows_kernel(const int* __restrict__ idx,
                                  const float* __restrict__ OTin, const float* __restrict__ OTout,
                                  const float* __restrict__ embed,
                                  const float** rowsK, const float** rowsQ) {
    int i = blockIdx.x * blockDim.x + threadIdx.x;
    if (i < B_ * 2 * KNN_) {
        int b = i / (2 * KNN_), n = i % (2 * KNN_);
        int r = idx[b * KNN_ + (n % KNN_)];
        rowsK[i] = (n < KNN_ ? OTin : OTout) + (size_t)r * P_;
    }
    if (i < 128) rowsQ[i] = embed + (size_t)((i < B_) ? i : 0) * P_;
}

// ---------------------------------------------------------------------------
// WMMA GEMM: C[M,2048] = A[M,24960] @ B[24960,2048], A rows via pointer table.
// f32->f16 convert during LDS staging, v_wmma_f32_16x16x32_f16 accumulate.
// Block = 256 thr = 8 waves; block tile 128(M) x 64(N); LDS K-stage = 64
// (2 K-substeps of 32 -> 8 WMMAs per barrier). Wave grid 4x2, wave = 32x32.
// N and K are compile-time so all address math is shifts (no v_mul_u64).
// ---------------------------------------------------------------------------
union AF { v16h v; v8h h[2]; };

__global__ __launch_bounds__(256)
void gemm_wmma_kernel(const float* const* __restrict__ Arows,
                      const float* __restrict__ B,
                      float* __restrict__ C,
                      int M) {
    __shared__ __align__(16) _Float16 ldsA[128 * 64];   // [m][k]  16 KB
    __shared__ __align__(16) _Float16 ldsB[64 * 64];    // [n][k]   8 KB (K-major)

    int tid  = threadIdx.x;
    int lane = tid & 31;
    int wave = tid >> 5;
    int waveM = wave & 3;       // 0..3
    int waveN = wave >> 2;      // 0..1
    int blockM = blockIdx.x * 128;
    int blockN = blockIdx.y * 64;

    v8f acc[2][2];
    acc[0][0] = {}; acc[0][1] = {}; acc[1][0] = {}; acc[1][1] = {};

    // A-load: thread -> (row, 32-float segment):  128 rows x 64 K
    int arow = tid >> 1;
    int aseg = (tid & 1) * 32;
    const float* abase = Arows[blockM + arow] + aseg;   // table padded to grid coverage
    _Float16* ldsA_dst = &ldsA[arow * 64 + aseg];

    // B-load: thread -> (k row, 16-col segment):  64 K x 64 N, store transposed
    int bk = tid >> 2;
    int bn = (tid & 3) * 16;

    int aKoff = (lane >= 16) ? 8 : 0;            // A per-lane K offset (halves)
    int bKoff = (lane >= 16) ? 8 : 0;            // B per-lane K offset (halves)

    const _Float16* paBase[2];
    const _Float16* pbBase[2];
    #pragma unroll
    for (int tm = 0; tm < 2; ++tm)
        paBase[tm] = &ldsA[(waveM * 32 + tm * 16 + (lane & 15)) * 64] + aKoff;
    #pragma unroll
    for (int tn = 0; tn < 2; ++tn)
        pbBase[tn] = &ldsB[(waveN * 32 + tn * 16 + (lane & 15)) * 64] + bKoff;

    for (int k0 = 0; k0 < P_; k0 += 64) {
        // --- stage A tile 128x64 (f32 -> f16) ---
        {
            const float* ap = abase + k0;
            #pragma unroll
            for (int i = 0; i < 32; i += 4) {
                float4 f = *(const float4*)(ap + i);
                ldsA_dst[i + 0] = (_Float16)f.x; ldsA_dst[i + 1] = (_Float16)f.y;
                ldsA_dst[i + 2] = (_Float16)f.z; ldsA_dst[i + 3] = (_Float16)f.w;
            }
        }
        // --- stage B tile 64x64, transposed to [n][k] (f32 -> f16) ---
        {
            const float* bp = B + (size_t)(k0 + bk) * DIM_ + blockN + bn;
            #pragma unroll
            for (int i = 0; i < 16; i += 4) {
                float4 f = *(const float4*)(bp + i);
                ldsB[(bn + i + 0) * 64 + bk] = (_Float16)f.x;
                ldsB[(bn + i + 1) * 64 + bk] = (_Float16)f.y;
                ldsB[(bn + i + 2) * 64 + bk] = (_Float16)f.z;
                ldsB[(bn + i + 3) * 64 + bk] = (_Float16)f.w;
            }
        }
        __syncthreads();

        // --- 2 K-substeps x (2x2 WMMA) = 8 WMMAs per barrier ---
        #pragma unroll
        for (int ks = 0; ks < 64; ks += 32) {
            AF a[2], b[2];
            #pragma unroll
            for (int tm = 0; tm < 2; ++tm) {
                const _Float16* pa = paBase[tm] + ks;
                a[tm].h[0] = *(const v8h*)(pa);
                a[tm].h[1] = *(const v8h*)(pa + 16);
            }
            #pragma unroll
            for (int tn = 0; tn < 2; ++tn) {
                const _Float16* pb = pbBase[tn] + ks;
                b[tn].h[0] = *(const v8h*)(pb);
                b[tn].h[1] = *(const v8h*)(pb + 16);
            }
            #pragma unroll
            for (int tm = 0; tm < 2; ++tm)
                #pragma unroll
                for (int tn = 0; tn < 2; ++tn)
                    acc[tm][tn] = __builtin_amdgcn_wmma_f32_16x16x32_f16(
                        false, a[tm].v, false, b[tn].v,
                        (short)0, acc[tm][tn], false, false);
        }
        __syncthreads();
    }

    // --- writeout: C/D layout  M = r + 8*(lane>=16), N = lane&15 ---
    // per-tile all-in/all-out guard; in-range tiles -> 8 stores w/ imm offsets
    int mhalf = (lane >= 16) ? 8 : 0;
    #pragma unroll
    for (int tm = 0; tm < 2; ++tm) {
        int mt = blockM + waveM * 32 + tm * 16;          // tile base row
        #pragma unroll
        for (int tn = 0; tn < 2; ++tn) {
            int nbase = blockN + waveN * 32 + tn * 16 + (lane & 15);
            float* cp = C + (size_t)(mt + mhalf) * DIM_ + nbase;
            if (mt + 16 <= M) {
                #pragma unroll
                for (int r = 0; r < 8; ++r)
                    cp[(size_t)r * DIM_] = acc[tm][tn][r];
            } else if (mt < M) {
                #pragma unroll
                for (int r = 0; r < 8; ++r)
                    if (mt + mhalf + r < M) cp[(size_t)r * DIM_] = acc[tm][tn][r];
            }
        }
    }
}

// ---------------------------------------------------------------------------
// Attention readout
// ---------------------------------------------------------------------------
__global__ void qk_kernel(const float* __restrict__ q, const float* __restrict__ kmat,
                          float* __restrict__ attn_s) {
    int i = blockIdx.x * blockDim.x + threadIdx.x;   // 64*20
    if (i >= B_ * 2 * KNN_) return;
    int b = i / (2 * KNN_);
    const float* qr = q + b * DIM_;
    const float* kr = kmat + (size_t)i * DIM_;
    float acc = 0.f;
    for (int d = 0; d < DIM_; ++d) acc += qr[d] * kr[d];
    attn_s[i] = acc * 0.02209708691f;   // 1/sqrt(2048)
}

__global__ void softmax_kernel(const float* __restrict__ attn_s, float* __restrict__ attn) {
    int b = threadIdx.x;
    if (b >= B_) return;
    const float* s = attn_s + b * 2 * KNN_;
    float m = -FLT_MAX;
    for (int n = 0; n < 2 * KNN_; ++n) m = fmaxf(m, s[n]);
    float sum = 0.f, e[2 * KNN_];
    for (int n = 0; n < 2 * KNN_; ++n) { e[n] = __expf(s[n] - m); sum += e[n]; }
    float inv = 1.0f / sum;
    for (int n = 0; n < 2 * KNN_; ++n) attn[b * 2 * KNN_ + n] = e[n] * inv;
}

__global__ void out_kernel(const float* __restrict__ attn, const float* const* __restrict__ rowsK,
                           float* __restrict__ out) {
    int p = blockIdx.x * blockDim.x + threadIdx.x;
    int b = blockIdx.y;
    if (p >= P_) return;
    float acc = 0.f;
    #pragma unroll
    for (int n = 0; n < 2 * KNN_; ++n)
        acc += attn[b * 2 * KNN_ + n] * rowsK[b * 2 * KNN_ + n][p];
    out[(size_t)b * P_ + p] = acc;
}

// ---------------------------------------------------------------------------
extern "C" void kernel_launch(void* const* d_in, const int* in_sizes, int n_in,
                              void* d_out, int out_size, void* d_ws, size_t ws_size,
                              hipStream_t stream) {
    const float* x       = (const float*)d_in[0];
    const float* dml     = (const float*)d_in[1];
    const float* OTin    = (const float*)d_in[2];
    const float* OTout   = (const float*)d_in[3];
    const float* W_dml   = (const float*)d_in[4];
    const float* W_patch = (const float*)d_in[5];
    const float* cls_tok = (const float*)d_in[6];
    const float* Wq      = (const float*)d_in[7];
    const float* Wk      = (const float*)d_in[8];
    float* out = (float*)d_out;

    char* base = (char*)d_ws;
    size_t off = 0;
    auto alloc = [&](size_t bytes) -> char* {
        off = (off + 255) & ~(size_t)255;
        char* p = base + off;
        off += bytes;
        return p;
    };
    float* cls    = (float*)alloc(B_ * DF_ * sizeof(float));
    float* dml2   = (float*)alloc(N_DML * sizeof(float));
    float* score  = (float*)alloc((size_t)B_ * N_DML * sizeof(float));
    int*   idx    = (int*)  alloc(B_ * KNN_ * sizeof(int));
    float* embed  = (float*)alloc((size_t)B_ * P_ * sizeof(float));
    float* q      = (float*)alloc((size_t)B_ * DIM_ * sizeof(float));
    float* kmat   = (float*)alloc((size_t)B_ * 2 * KNN_ * DIM_ * sizeof(float));
    float* attn_s = (float*)alloc(B_ * 2 * KNN_ * sizeof(float));
    float* attn   = (float*)alloc(B_ * 2 * KNN_ * sizeof(float));
    const float** rowsK = (const float**)alloc((B_ * 2 * KNN_) * sizeof(float*));
    const float** rowsQ = (const float**)alloc(128 * sizeof(float*));

    // 1) retrieval path (f32)
    cls_kernel<<<dim3((B_ * DF_ + 255) / 256), 256, 0, stream>>>(x, W_dml, cls);
    dml2_kernel<<<dim3((N_DML + 255) / 256), 256, 0, stream>>>(dml, dml2);
    score_kernel<<<dim3((N_DML + 255) / 256, B_), 256, 0, stream>>>(cls, dml, dml2, score);
    topk_kernel<<<dim3(B_), 256, 0, stream>>>(score, idx);

    // 2) patch embed
    patch_embed_kernel<<<dim3((B_ * TOK_ * HID_ + 255) / 256), 256, 0, stream>>>(
        x, W_patch, cls_tok, embed);

    // 3) gather-by-indirection tables
    build_rows_kernel<<<dim3((B_ * 2 * KNN_ + 255) / 256), 256, 0, stream>>>(
        idx, OTin, OTout, embed, rowsK, rowsQ);

    // 4) WMMA GEMMs: q = embed@Wq  (M=64 -> 1 M-block), k = neighbors@Wk (M=1280)
    gemm_wmma_kernel<<<dim3(1, DIM_ / 64), 256, 0, stream>>>(rowsQ, Wq, q, B_);
    gemm_wmma_kernel<<<dim3((B_ * 2 * KNN_) / 128, DIM_ / 64), 256, 0, stream>>>(
        rowsK, Wk, kmat, B_ * 2 * KNN_);

    // 5) attention readout
    qk_kernel<<<dim3((B_ * 2 * KNN_ + 255) / 256), 256, 0, stream>>>(q, kmat, attn_s);
    softmax_kernel<<<dim3(1), 64, 0, stream>>>(attn_s, attn);
    out_kernel<<<dim3((P_ + 255) / 256, B_), 256, 0, stream>>>(attn, rowsK, out);
}
